// DomainGCN_712964571494
// MI455X (gfx1250) — matrix operand order repaired and checked by reading, not compiled, not run on hardware
//
#include <hip/hip_runtime.h>
#include <hip/hip_bf16.h>

typedef __attribute__((ext_vector_type(16))) __bf16        v16bf;
typedef __attribute__((ext_vector_type(8)))  float         v8f;
typedef __attribute__((ext_vector_type(4)))  unsigned int  v4u;

#define NN        4096      // nodes
#define DF        256       // feature dim
#define NHEADS    4
#define DKH       64
#define NSK       8192      // skills
#define NEG_SLOPE 0.2f
#define LN_EPS    1e-5f

// ---------------------------------------------------------------------------
// CDNA5 async global->LDS copy (16B per lane), tracked by ASYNCcnt.
// ---------------------------------------------------------------------------
static __device__ __forceinline__ void async_ld16(unsigned lds_off, const void* g) {
    unsigned long long ga = (unsigned long long)(uintptr_t)g;
    asm volatile("global_load_async_to_lds_b128 %0, %1, off"
                 :: "v"(lds_off), "v"(ga) : "memory");
}
static __device__ __forceinline__ void wait_async4() {
    asm volatile("s_wait_asynccnt 0x4" ::: "memory");
}
static __device__ __forceinline__ void wait_async0() {
    asm volatile("s_wait_asynccnt 0x0" ::: "memory");
}

// ---------------------------------------------------------------------------
// LDS fragment loaders matching CDNA5 wave32 WMMA VGPR layouts (16x16x32 bf16)
//   A tile in LDS: row-major [128][32] bf16
//   B tile in LDS: [N][K] (transposed) [64][32] bf16 -> per-lane contiguous
// ---------------------------------------------------------------------------
static __device__ __forceinline__ v16bf frag_a_lds(const __bf16* lA, int m0) {
    const int lane = threadIdx.x & 31;
    const __bf16* row = lA + (size_t)(m0 + (lane & 15)) * 32;
    const int koff = (lane < 16) ? 0 : 8;
    union { v4u u[2]; v16bf b; } cv;
    cv.u[0] = *(const v4u*)(row + koff);
    cv.u[1] = *(const v4u*)(row + 16 + koff);
    return cv.b;
}

static __device__ __forceinline__ v16bf frag_b_lds(const __bf16* lB, int n0) {
    const int lane = threadIdx.x & 31;
    const __bf16* col = lB + (size_t)(n0 + (lane & 15)) * 32 + ((lane < 16) ? 0 : 8);
    union { v4u u[2]; v16bf b; } cv;
    cv.u[0] = *(const v4u*)(col);
    cv.u[1] = *(const v4u*)(col + 16);
    return cv.b;
}

#define WMMA_BF16(A, B, C) __builtin_amdgcn_wmma_f32_16x16x32_bf16( \
        false, (A), false, (B), (short)0, (C), false, false)

// ---------------------------------------------------------------------------
// GEMM 1: C[i,j] = act( sum_k X[i,k] * W[j,k] + bias[j] )   (X:[M][K], W:[N][K])
// grid: (N/64, M/128), block 256.  K small (256) -> direct loads, no staging.
// ---------------------------------------------------------------------------
__global__ __launch_bounds__(256) void gemm_xwT_kernel(
    const float* __restrict__ X, const float* __restrict__ W,
    const float* __restrict__ bias, float* __restrict__ C,
    int M, int K, int ldC, int relu)
{
    __shared__ __attribute__((aligned(16))) __bf16 lA[128 * 32];
    __shared__ __attribute__((aligned(16))) __bf16 lB[64 * 32];
    const int tid = threadIdx.x;
    const int i0 = blockIdx.y * 128;
    const int j0 = blockIdx.x * 64;
    const int m0 = (tid >> 5) * 16;

    v8f acc0 = {}, acc1 = {}, acc2 = {}, acc3 = {};
    for (int k0 = 0; k0 < K; k0 += 32) {
#pragma unroll
        for (int it = 0; it < 16; ++it) {          // A tile: 128x32
            int e = tid + it * 256;
            int r = e >> 5, c = e & 31;
            lA[e] = (__bf16)X[(size_t)(i0 + r) * K + k0 + c];
        }
#pragma unroll
        for (int it = 0; it < 8; ++it) {           // B tile: lB[n][k] = W[j0+n][k0+k]
            int e = tid + it * 256;
            int n = e >> 5, c = e & 31;
            lB[n * 32 + c] = (__bf16)W[(size_t)(j0 + n) * K + k0 + c];
        }
        __syncthreads();
        v16bf a  = frag_a_lds(lA, m0);
        v16bf b0 = frag_b_lds(lB, 0);
        v16bf b1 = frag_b_lds(lB, 16);
        v16bf b2 = frag_b_lds(lB, 32);
        v16bf b3 = frag_b_lds(lB, 48);
        acc0 = WMMA_BF16(a, b0, acc0);
        acc1 = WMMA_BF16(a, b1, acc1);
        acc2 = WMMA_BF16(a, b2, acc2);
        acc3 = WMMA_BF16(a, b3, acc3);
        __syncthreads();
    }
    const int lane = tid & 31;
    const int ncol = j0 + (lane & 15);
    const int rbase = i0 + m0 + ((lane < 16) ? 0 : 8);
    v8f accs[4] = {acc0, acc1, acc2, acc3};
#pragma unroll
    for (int t = 0; t < 4; ++t) {
        int col = ncol + t * 16;
        float bv = bias ? bias[col] : 0.f;
#pragma unroll
        for (int r = 0; r < 8; ++r) {
            float v = accs[t][r] + bv;
            if (relu) v = fmaxf(v, 0.f);
            C[(size_t)(rbase + r) * ldC + col] = v;
        }
    }
}

// ---------------------------------------------------------------------------
// GEMM 2: C = act( (rowscale .* (A + diag*I) .* colscale) @ B )
//   A:[M][K] f32 adjacency (async-staged to LDS, double buffered),
//   B:[K][ldB] f32 row-major.
// grid: (Nb/64, M/128)
// ---------------------------------------------------------------------------
__global__ __launch_bounds__(256) void gemm_scaled_adj_kernel(
    const float* __restrict__ A, const float* __restrict__ rowscale,
    const float* __restrict__ colscale, const float* __restrict__ B,
    float* __restrict__ C, int M, int K, int ldB, int ldC,
    int add_diag, int relu)
{
    __shared__ __attribute__((aligned(16))) float  sA[2][128 * 32];  // f32 staging
    __shared__ __attribute__((aligned(16))) __bf16 lA[128 * 32];
    __shared__ __attribute__((aligned(16))) __bf16 lB[64 * 32];
    __shared__ float sRow[128];
    const int tid = threadIdx.x;
    const int i0 = blockIdx.y * 128;
    const int j0 = blockIdx.x * 64;
    const int m0 = (tid >> 5) * 16;

    const unsigned base0 = (unsigned)(uintptr_t)&sA[0][0];
    const unsigned base1 = (unsigned)(uintptr_t)&sA[1][0];

    // Each thread stages 4 x 16B chunks of the 128x32 f32 A tile.
    auto issueA = [&](unsigned base, int k0n) {
#pragma unroll
        for (int it = 0; it < 4; ++it) {
            int q = tid + it * 256;
            int r = q >> 3, c = (q & 7) << 2;
            async_ld16(base + (unsigned)q * 16u,
                       A + (size_t)(i0 + r) * K + (k0n + c));
        }
    };

    v8f acc0 = {}, acc1 = {}, acc2 = {}, acc3 = {};
    issueA(base0, 0);
    if (tid < 128) sRow[tid] = rowscale[i0 + tid];
    __syncthreads();

    int ib = 0;
    for (int k0 = 0; k0 < K; k0 += 32, ib ^= 1) {
        const bool pre = (k0 + 32 < K);
        if (pre) issueA(ib ? base0 : base1, k0 + 32);
#pragma unroll
        for (int it = 0; it < 8; ++it) {           // lB[n][k] = B[k0+k][j0+n]
            int e = tid + it * 256;
            int k = e >> 6, n = e & 63;
            lB[n * 32 + k] = (__bf16)B[(size_t)(k0 + k) * ldB + j0 + n];
        }
        if (pre) wait_async4(); else wait_async0();
        {   // convert own staged chunks f32 -> scaled bf16
            const float* st = ib ? &sA[1][0] : &sA[0][0];
#pragma unroll
            for (int it = 0; it < 4; ++it) {
                int q = tid + it * 256;
#pragma unroll
                for (int j = 0; j < 4; ++j) {
                    int f = q * 4 + j;
                    int r = f >> 5, c = f & 31;
                    int gi = i0 + r, gk = k0 + c;
                    float v = st[f];
                    if (add_diag && gi == gk) v += 1.f;
                    v *= sRow[r];
                    if (colscale) v *= colscale[gk];
                    lA[f] = (__bf16)v;
                }
            }
        }
        __syncthreads();
        v16bf a  = frag_a_lds(lA, m0);
        v16bf b0 = frag_b_lds(lB, 0);
        v16bf b1 = frag_b_lds(lB, 16);
        v16bf b2 = frag_b_lds(lB, 32);
        v16bf b3 = frag_b_lds(lB, 48);
        acc0 = WMMA_BF16(a, b0, acc0);
        acc1 = WMMA_BF16(a, b1, acc1);
        acc2 = WMMA_BF16(a, b2, acc2);
        acc3 = WMMA_BF16(a, b3, acc3);
        __syncthreads();
    }
    const int lane = tid & 31;
    const int ncol = j0 + (lane & 15);
    const int rbase = i0 + m0 + ((lane < 16) ? 0 : 8);
    v8f accs[4] = {acc0, acc1, acc2, acc3};
#pragma unroll
    for (int t = 0; t < 4; ++t) {
#pragma unroll
        for (int r = 0; r < 8; ++r) {
            float v = accs[t][r];
            if (relu) v = fmaxf(v, 0.f);
            C[(size_t)(rbase + r) * ldC + ncol + t * 16] = v;
        }
    }
}

// ---------------------------------------------------------------------------
// GEMM 3: attention aggregation, one head per blockIdx.z
//   alpha[i,k] = mask ? exp(leaky(sl[i,h]+sr[k,h]) - m[i,h]) * invZ[i,h] : 0
//   agg[i, h*64+j] = sum_k alpha[i,k] * Wh[k, h*64+j]
// grid: (1, NN/128, 4); A_dom async-staged, alpha materialized per tile.
// ---------------------------------------------------------------------------
__global__ __launch_bounds__(256) void gemm_attn_kernel(
    const float* __restrict__ Adom, const float* __restrict__ sl,
    const float* __restrict__ sr, const float* __restrict__ mrow,
    const float* __restrict__ invZ, const float* __restrict__ Wh,
    float* __restrict__ agg)
{
    __shared__ __attribute__((aligned(16))) float  sA[2][128 * 32];
    __shared__ __attribute__((aligned(16))) __bf16 lA[128 * 32];
    __shared__ __attribute__((aligned(16))) __bf16 lB[64 * 32];
    __shared__ float sSl[128], sM[128], sIz[128];
    const int tid = threadIdx.x;
    const int h = blockIdx.z;
    const int i0 = blockIdx.y * 128;
    const int m0 = (tid >> 5) * 16;

    const unsigned base0 = (unsigned)(uintptr_t)&sA[0][0];
    const unsigned base1 = (unsigned)(uintptr_t)&sA[1][0];

    auto issueA = [&](unsigned base, int k0n) {
#pragma unroll
        for (int it = 0; it < 4; ++it) {
            int q = tid + it * 256;
            int r = q >> 3, c = (q & 7) << 2;
            async_ld16(base + (unsigned)q * 16u,
                       Adom + (size_t)(i0 + r) * NN + (k0n + c));
        }
    };

    v8f acc0 = {}, acc1 = {}, acc2 = {}, acc3 = {};
    issueA(base0, 0);
    if (tid < 128) {
        int gi = i0 + tid;
        sSl[tid] = sl[gi * NHEADS + h];
        sM[tid]  = mrow[gi * NHEADS + h];
        sIz[tid] = invZ[gi * NHEADS + h];
    }
    __syncthreads();

    int ib = 0;
    for (int k0 = 0; k0 < NN; k0 += 32, ib ^= 1) {
        const bool pre = (k0 + 32 < NN);
        if (pre) issueA(ib ? base0 : base1, k0 + 32);
#pragma unroll
        for (int it = 0; it < 8; ++it) {           // lB[n][k] = Wh[k0+k][h*64+n]
            int e = tid + it * 256;
            int k = e >> 6, n = e & 63;
            lB[n * 32 + k] = (__bf16)Wh[(size_t)(k0 + k) * DF + h * DKH + n];
        }
        if (pre) wait_async4(); else wait_async0();
        {   // materialize alpha tile from own staged chunks
            const float* st = ib ? &sA[1][0] : &sA[0][0];
#pragma unroll
            for (int it = 0; it < 4; ++it) {
                int q = tid + it * 256;
#pragma unroll
                for (int j = 0; j < 4; ++j) {
                    int f = q * 4 + j;
                    int r = f >> 5, c = f & 31;
                    int gi = i0 + r, gk = k0 + c;
                    float a = st[f];
                    float v = 0.f;
                    if (a > 0.f || gi == gk) {
                        float s = sSl[r] + sr[gk * NHEADS + h];
                        s = (s > 0.f) ? s : NEG_SLOPE * s;
                        v = __expf(s - sM[r]) * sIz[r];
                    }
                    lA[f] = (__bf16)v;
                }
            }
        }
        __syncthreads();
        v16bf a  = frag_a_lds(lA, m0);
        v16bf b0 = frag_b_lds(lB, 0);
        v16bf b1 = frag_b_lds(lB, 16);
        v16bf b2 = frag_b_lds(lB, 32);
        v16bf b3 = frag_b_lds(lB, 48);
        acc0 = WMMA_BF16(a, b0, acc0);
        acc1 = WMMA_BF16(a, b1, acc1);
        acc2 = WMMA_BF16(a, b2, acc2);
        acc3 = WMMA_BF16(a, b3, acc3);
        __syncthreads();
    }
    const int lane = tid & 31;
    const int ncol = h * DKH + (lane & 15);
    const int rbase = i0 + m0 + ((lane < 16) ? 0 : 8);
    v8f accs[4] = {acc0, acc1, acc2, acc3};
#pragma unroll
    for (int t = 0; t < 4; ++t) {
#pragma unroll
        for (int r = 0; r < 8; ++r) {
            agg[(size_t)(rbase + r) * DF + ncol + t * 16] = accs[t][r];
        }
    }
}

// ---------------------------------------------------------------------------
// d_inv[i] = (rowsum(A_dom[i,:]) + 1)^-1/2
// ---------------------------------------------------------------------------
__global__ __launch_bounds__(256) void dinv_kernel(const float* __restrict__ A,
                                                   float* __restrict__ dinv)
{
    const int i = blockIdx.x, t = threadIdx.x;
    __shared__ float red[256];
    float s = 0.f;
    for (int j = t; j < NN; j += 256) s += A[(size_t)i * NN + j];
    red[t] = s; __syncthreads();
    for (int st = 128; st > 0; st >>= 1) {
        if (t < st) red[t] += red[t + st];
        __syncthreads();
    }
    if (t == 0) dinv[i] = rsqrtf(red[0] + 1.f);
}

// inv_deg_s[i] = 1 / max(rowsum(A_ds[i,:]), 1)
__global__ __launch_bounds__(256) void inv_degs_kernel(const float* __restrict__ A,
                                                       float* __restrict__ inv)
{
    const int i = blockIdx.x, t = threadIdx.x;
    __shared__ float red[256];
    float s = 0.f;
    for (int j = t; j < NN; j += 256) s += A[(size_t)i * NN + j];
    red[t] = s; __syncthreads();
    for (int st = 128; st > 0; st >>= 1) {
        if (t < st) red[t] += red[t + st];
        __syncthreads();
    }
    if (t == 0) inv[i] = 1.f / fmaxf(red[0], 1.f);
}

// ---------------------------------------------------------------------------
// sl[i,h] = dot(Wh[i, h*64:(h+1)*64], a_l[h,:]) ; sr likewise
// ---------------------------------------------------------------------------
__global__ __launch_bounds__(256) void slsr_kernel(
    const float* __restrict__ Wh, const float* __restrict__ al,
    const float* __restrict__ ar, float* __restrict__ sl, float* __restrict__ sr)
{
    const int i = blockIdx.x, t = threadIdx.x;
    __shared__ float rl[256], rr[256];
    float w = Wh[(size_t)i * DF + t];
    rl[t] = w * al[t];
    rr[t] = w * ar[t];
    __syncthreads();
    for (int st = 32; st > 0; st >>= 1) {
        if ((t & 63) < st) { rl[t] += rl[t + st]; rr[t] += rr[t + st]; }
        __syncthreads();
    }
    if ((t & 63) == 0) {
        int h = t >> 6;
        sl[i * NHEADS + h] = rl[t];
        sr[i * NHEADS + h] = rr[t];
    }
}

// ---------------------------------------------------------------------------
// per-row masked softmax statistics: mrow[i,h] = max, invZ[i,h] = 1/sum(exp)
// ---------------------------------------------------------------------------
__global__ __launch_bounds__(256) void attn_mz_kernel(
    const float* __restrict__ Adom, const float* __restrict__ sl,
    const float* __restrict__ sr, float* __restrict__ mrow,
    float* __restrict__ invZ)
{
    const int i = blockIdx.x, t = threadIdx.x;
    __shared__ float red[256];
    float slh[NHEADS];
#pragma unroll
    for (int h = 0; h < NHEADS; ++h) slh[h] = sl[i * NHEADS + h];

    float mx[NHEADS] = {-3.0e38f, -3.0e38f, -3.0e38f, -3.0e38f};
    for (int j = t; j < NN; j += 256) {
        float a = Adom[(size_t)i * NN + j];
        if (a > 0.f || j == i) {
#pragma unroll
            for (int h = 0; h < NHEADS; ++h) {
                float s = slh[h] + sr[j * NHEADS + h];
                s = (s > 0.f) ? s : NEG_SLOPE * s;
                mx[h] = fmaxf(mx[h], s);
            }
        }
    }
    float mfin[NHEADS];
#pragma unroll
    for (int h = 0; h < NHEADS; ++h) {
        __syncthreads();
        red[t] = mx[h];
        __syncthreads();
        for (int st = 128; st > 0; st >>= 1) {
            if (t < st) red[t] = fmaxf(red[t], red[t + st]);
            __syncthreads();
        }
        mfin[h] = red[0];
    }
    float sum[NHEADS] = {0.f, 0.f, 0.f, 0.f};
    for (int j = t; j < NN; j += 256) {
        float a = Adom[(size_t)i * NN + j];
        if (a > 0.f || j == i) {
#pragma unroll
            for (int h = 0; h < NHEADS; ++h) {
                float s = slh[h] + sr[j * NHEADS + h];
                s = (s > 0.f) ? s : NEG_SLOPE * s;
                sum[h] += __expf(s - mfin[h]);
            }
        }
    }
#pragma unroll
    for (int h = 0; h < NHEADS; ++h) {
        __syncthreads();
        red[t] = sum[h];
        __syncthreads();
        for (int st = 128; st > 0; st >>= 1) {
            if (t < st) red[t] += red[t + st];
            __syncthreads();
        }
        if (t == 0) {
            mrow[i * NHEADS + h] = mfin[h];
            invZ[i * NHEADS + h] = 1.f / red[0];
        }
    }
}

// ---------------------------------------------------------------------------
// x = elu(agg) + H_dom ; LayerNorm(x) * gamma + beta  (one row per block)
// ---------------------------------------------------------------------------
__global__ __launch_bounds__(256) void resid_ln_kernel(
    const float* __restrict__ agg, const float* __restrict__ Hdom,
    const float* __restrict__ gamma, const float* __restrict__ beta,
    float* __restrict__ xln)
{
    const int i = blockIdx.x, t = threadIdx.x;
    __shared__ float red[256];
    float v = agg[(size_t)i * DF + t];
    float hh = (v > 0.f) ? v : (__expf(v) - 1.f);
    float x = hh + Hdom[(size_t)i * DF + t];

    red[t] = x; __syncthreads();
    for (int st = 128; st > 0; st >>= 1) {
        if (t < st) red[t] += red[t + st];
        __syncthreads();
    }
    float mu = red[0] * (1.f / DF);
    __syncthreads();
    float d = x - mu;
    red[t] = d * d; __syncthreads();
    for (int st = 128; st > 0; st >>= 1) {
        if (t < st) red[t] += red[t + st];
        __syncthreads();
    }
    float var = red[0] * (1.f / DF);
    xln[(size_t)i * DF + t] = d * rsqrtf(var + LN_EPS) * gamma[t] + beta[t];
}

// ---------------------------------------------------------------------------
extern "C" void kernel_launch(void* const* d_in, const int* in_sizes, int n_in,
                              void* d_out, int out_size, void* d_ws, size_t ws_size,
                              hipStream_t stream)
{
    (void)in_sizes; (void)n_in; (void)out_size; (void)ws_size;
    const float* H_dom   = (const float*)d_in[0];
    const float* A_dom   = (const float*)d_in[1];
    const float* A_ds    = (const float*)d_in[2];
    const float* W_basic = (const float*)d_in[3];
    const float* b_basic = (const float*)d_in[4];
    const float* W_attn  = (const float*)d_in[5];
    const float* a_l     = (const float*)d_in[6];
    const float* a_r     = (const float*)d_in[7];
    const float* ln_g    = (const float*)d_in[8];
    const float* ln_b    = (const float*)d_in[9];
    const float* W_d2s   = (const float*)d_in[10];
    const float* b_d2s   = (const float*)d_in[11];
    float* out = (float*)d_out;

    char* ws = (char*)d_ws;
    size_t off = 0;
    auto alloc = [&](size_t nf) { float* p = (float*)(ws + off); off += nf * sizeof(float); return p; };
    float* dinv  = alloc(NN);
    float* idegs = alloc(NSK);
    float* sl    = alloc((size_t)NN * NHEADS);
    float* sr    = alloc((size_t)NN * NHEADS);
    float* mrow  = alloc((size_t)NN * NHEADS);
    float* invZ  = alloc((size_t)NN * NHEADS);
    float* bufA  = alloc((size_t)NN * DF);   // T1, later WH
    float* bufB  = alloc((size_t)NN * DF);   // h0, later x_ln
    float* bufC  = alloc((size_t)NN * DF);   // Wh
    float* bufD  = alloc((size_t)NN * DF);   // agg

    dim3 blk(256);

    // 1) degree norm of (A_dom + I)
    dinv_kernel<<<NN, blk, 0, stream>>>(A_dom, dinv);
    // 2) T1 = H_dom @ W_basic^T + b_basic
    gemm_xwT_kernel<<<dim3(DF / 64, NN / 128), blk, 0, stream>>>(
        H_dom, W_basic, b_basic, bufA, NN, DF, DF, 0);
    // 3) h0 = relu(A_norm @ T1)
    gemm_scaled_adj_kernel<<<dim3(DF / 64, NN / 128), blk, 0, stream>>>(
        A_dom, dinv, dinv, bufA, bufB, NN, NN, DF, DF, 1, 1);
    // 4) Wh = h0 @ W_attn^T
    gemm_xwT_kernel<<<dim3(DF / 64, NN / 128), blk, 0, stream>>>(
        bufB, W_attn, nullptr, bufC, NN, DF, DF, 0);
    // 5) attention scores
    slsr_kernel<<<NN, blk, 0, stream>>>(bufC, a_l, a_r, sl, sr);
    // 6) masked softmax stats (max, 1/Z)
    attn_mz_kernel<<<NN, blk, 0, stream>>>(A_dom, sl, sr, mrow, invZ);
    // 7) agg = alpha @ Wh per head
    gemm_attn_kernel<<<dim3(1, NN / 128, NHEADS), blk, 0, stream>>>(
        A_dom, sl, sr, mrow, invZ, bufC, bufD);
    // 8) x_ln = LN(elu(agg) + H_dom)
    resid_ln_kernel<<<NN, blk, 0, stream>>>(bufD, H_dom, ln_g, ln_b, bufB);
    // 9) WH = x_ln @ W_d2s^T + b_d2s
    gemm_xwT_kernel<<<dim3(DF / 64, NN / 128), blk, 0, stream>>>(
        bufB, W_d2s, b_d2s, bufA, NN, DF, DF, 0);
    // 10) bipartite degrees
    inv_degs_kernel<<<NSK, blk, 0, stream>>>(A_ds, idegs);
    // 11) out = relu((A_ds / deg_s) @ WH)
    gemm_scaled_adj_kernel<<<dim3(DF / 64, NSK / 128), blk, 0, stream>>>(
        A_ds, idegs, nullptr, bufA, out, NSK, NN, DF, DF, 0, 1);
}
